// seq2seqBase_44702019617253
// MI455X (gfx1250) — matrix operand order repaired and checked by reading, not compiled, not run on hardware
//
#include <hip/hip_runtime.h>
#include <hip/hip_bf16.h>
#include <math.h>

// ---------------- constants ----------------
#define BB    32
#define SSRC  128
#define STGT  64
#define HH    300
#define VV    32000
#define HP    304          // N padded to multiple of 16
#define KP    320          // K padded to multiple of 32
#define MENC  (BB*SSRC)    // 4096
#define MDEC  (BB*STGT)    // 2048

typedef __bf16 bf16;
typedef __attribute__((ext_vector_type(16))) __bf16 v16bf;
typedef __attribute__((ext_vector_type(8)))  __bf16 v8bf;
typedef __attribute__((ext_vector_type(8)))  float  v8f;

// ---------------- helpers ----------------
__device__ __forceinline__ bf16 f2bf(float f) {
  union { float f; unsigned u; } x; x.f = f;
  unsigned r = x.u + 0x7FFFu + ((x.u >> 16) & 1u);   // round-to-nearest-even
  unsigned short h = (unsigned short)(r >> 16);
  return __builtin_bit_cast(bf16, h);
}

__device__ __forceinline__ v8f zero8() {
  v8f a;
#pragma unroll
  for (int i = 0; i < 8; ++i) a[i] = 0.0f;
  return a;
}

// A fragment: 16x32 bf16 tile, row-major source, lda in elements.
// lane 0-15: row m=lane, K 0..15; lane 16-31: row m=lane-16, K 16..31
__device__ __forceinline__ v16bf load_a_frag(const bf16* base, int lda, int lane) {
  int m  = lane & 15;
  int k0 = (lane >> 4) << 4;                 // 0 or 16
  return *(const v16bf*)(base + m * lda + k0);
}

// B fragment: 32x16 (KxN); source is W row-major [N][K] (B = W^T).
// lane 0-15: col n=lane, K kk..kk+15; lane 16-31: col n=lane-16, K kk+16..kk+31
__device__ __forceinline__ v16bf load_b_frag(const bf16* base, int ldb, int lane) {
  int n  = lane & 15;
  int kh = (lane >> 4) << 4;                 // 0 or 16
  return *(const v16bf*)(base + n * ldb + kh);
}

__device__ __forceinline__ v8f wmma_bf16(v16bf a, v16bf b, v8f c) {
  return __builtin_amdgcn_wmma_f32_16x16x32_bf16(false, a, false, b, (short)0, c,
                                                 false, false);
}

// ---------------- kernels ----------------

// f32 [rows_valid][HH] -> bf16 [rows_out][KP], zero padded
__global__ void k_convert_weight(const float* __restrict__ W, bf16* __restrict__ out,
                                 int rows_valid, int rows_out) {
  long long idx = (long long)blockIdx.x * blockDim.x + threadIdx.x;
  long long total = (long long)rows_out * KP;
  if (idx >= total) return;
  int r = (int)(idx / KP), k = (int)(idx % KP);
  float v = (r < rows_valid && k < HH) ? W[(long long)r * HH + k] : 0.0f;
  out[idx] = f2bf(v);
}

// gather rows of E via ids -> bf16 [nrows][KP], zero padded
__global__ void k_gather_embed(const int* __restrict__ ids, const float* __restrict__ E,
                               bf16* __restrict__ out, int nrows) {
  long long idx = (long long)blockIdx.x * blockDim.x + threadIdx.x;
  long long total = (long long)nrows * KP;
  if (idx >= total) return;
  int r = (int)(idx / KP), k = (int)(idx % KP);
  float v = 0.0f;
  if (k < HH) v = E[(long long)ids[r] * HH + k];
  out[idx] = f2bf(v);
}

// Xenc[t][b][n] = (EN[x] @ Wih^T)[b*128+t][n] + bih[n] + bhh[n]   (f32, [128][32][HP])
__global__ void __launch_bounds__(256)
k_gemm_enc_x(const bf16* __restrict__ A, const bf16* __restrict__ Wb,
             const float* __restrict__ bih, const float* __restrict__ bhh,
             float* __restrict__ Xenc) {
  int wid = threadIdx.x >> 5, lane = threadIdx.x & 31;
  int tile = blockIdx.x * 8 + wid;           // 4864 tiles exactly
  int mt = tile / (HP / 16), nt = tile % (HP / 16);
  int m0 = mt * 16, n0 = nt * 16;
  v8f acc = zero8();
#pragma unroll
  for (int kk = 0; kk < KP; kk += 32) {
    v16bf a = load_a_frag(A + (long long)m0 * KP + kk, KP, lane);
    v16bf b = load_b_frag(Wb + (long long)n0 * KP + kk, KP, lane);
    acc = wmma_bf16(a, b, acc);
  }
  int col = n0 + (lane & 15);
  float bias = (col < HH) ? (bih[col] + bhh[col]) : 0.0f;
  int half8 = (lane >> 4) << 3;
#pragma unroll
  for (int r = 0; r < 8; ++r) {
    int row = m0 + r + half8;                // row = b*128 + t
    int bi = row >> 7, ti = row & 127;
    Xenc[(long long)(ti * 32 + bi) * HP + col] = acc[r] + bias;
  }
}

// persistent 1-block encoder scan: h = tanh(Xenc[t] + h @ Whh^T), 128 steps
__global__ void __launch_bounds__(256)
k_encoder_scan(const bf16* __restrict__ Whh, const float* __restrict__ h0,
               const float* __restrict__ Xenc, float* __restrict__ c) {
  __shared__ __align__(128) bf16 hbuf[2][32 * KP];
  int wid = threadIdx.x >> 5, lane = threadIdx.x & 31;
  for (int i = threadIdx.x; i < 32 * KP; i += 256) {
    int m = i / KP, k = i % KP;
    hbuf[0][i] = (k < HH) ? f2bf(h0[m * HH + k]) : f2bf(0.0f);
    hbuf[1][i] = f2bf(0.0f);
  }
  __syncthreads();
  int cur = 0;
  for (int t = 0; t < SSRC; ++t) {
    const bf16* hc = hbuf[cur];
    bf16* hn = hbuf[cur ^ 1];
    for (int tile = wid; tile < 2 * (HP / 16); tile += 8) {   // 38 tiles, wave-uniform
      int mt = tile / (HP / 16), nt = tile % (HP / 16);
      int m0 = mt * 16, n0 = nt * 16;
      v8f acc = zero8();
#pragma unroll
      for (int kk = 0; kk < KP; kk += 32) {
        v16bf a = load_a_frag(hc + m0 * KP + kk, KP, lane);
        v16bf b = load_b_frag(Whh + (long long)n0 * KP + kk, KP, lane);
        acc = wmma_bf16(a, b, acc);
      }
      int col = n0 + (lane & 15);
      int half8 = (lane >> 4) << 3;
#pragma unroll
      for (int r = 0; r < 8; ++r) {
        int row = m0 + r + half8;
        float v = tanhf(acc[r] + Xenc[(long long)(t * 32 + row) * HP + col]);
        hn[row * KP + col] = f2bf(v);
        if (t == SSRC - 1) c[row * HP + col] = v;
      }
    }
    __syncthreads();
    cur ^= 1;
  }
}

// term[b][n] = c[b] . dWhh[n] + dbhh[n] + dbih[n]   (f32 [32][HP], pad cols = 0)
__global__ void k_dec_term(const float* __restrict__ c, const float* __restrict__ dWhh,
                           const float* __restrict__ dbhh, const float* __restrict__ dbih,
                           float* __restrict__ term) {
  int idx = blockIdx.x * blockDim.x + threadIdx.x;
  if (idx >= 32 * HP) return;
  int b = idx / HP, n = idx % HP;
  float s = 0.0f;
  if (n < HH) {
    s = dbhh[n] + dbih[n];
    for (int k = 0; k < HH; ++k) s += c[b * HP + k] * dWhh[(long long)n * HH + k];
  }
  term[idx] = s;
}

// Hdec = tanh(ZH[y] @ dWih^T + term[b])  -> bf16 [2048][KP]
__global__ void __launch_bounds__(256)
k_gemm_dec_h(const bf16* __restrict__ A, const bf16* __restrict__ Wb,
             const float* __restrict__ term, bf16* __restrict__ Hdec) {
  int wid = threadIdx.x >> 5, lane = threadIdx.x & 31;
  int tile = blockIdx.x * 8 + wid;           // 2432 tiles exactly
  int mt = tile / (HP / 16), nt = tile % (HP / 16);
  int m0 = mt * 16, n0 = nt * 16;
  v8f acc = zero8();
#pragma unroll
  for (int kk = 0; kk < KP; kk += 32) {
    v16bf a = load_a_frag(A + (long long)m0 * KP + kk, KP, lane);
    v16bf b = load_b_frag(Wb + (long long)n0 * KP + kk, KP, lane);
    acc = wmma_bf16(a, b, acc);
  }
  int col = n0 + (lane & 15);
  int half8 = (lane >> 4) << 3;
#pragma unroll
  for (int r = 0; r < 8; ++r) {
    int row = m0 + r + half8;                // row = b*64 + t
    int bi = row >> 6;
    float v = tanhf(acc[r] + term[bi * HP + col]);
    Hdec[(long long)row * KP + col] = f2bf(v);
  }
}

// zero K-pad cols 304..319 of Hdec
__global__ void k_pad_hdec(bf16* __restrict__ Hdec) {
  int idx = blockIdx.x * blockDim.x + threadIdx.x;
  if (idx >= MDEC * (KP - HP)) return;
  int row = idx / (KP - HP), k = HP + idx % (KP - HP);
  Hdec[(long long)row * KP + k] = f2bf(0.0f);
}

// out[b,t,v] = Hdec @ Wout^T + bout    f32 [2048][32000]
// Register-blocked: each wave computes a 16M x 64N strip (1 A-frag reused over
// 4 B-frags / 4 accumulators).  Output streamed with non-temporal stores so the
// 262MB result does not evict the 20MB bf16 W_out from the 192MB L2.
__global__ void __launch_bounds__(256)
k_gemm_logits(const bf16* __restrict__ Hd, const bf16* __restrict__ Wo,
              const float* __restrict__ bout, float* __restrict__ out) {
  int wid = threadIdx.x >> 5, lane = threadIdx.x & 31;
  int m0 = (blockIdx.y * 2 + (wid >> 2)) * 16;     // < 2048
  int n0 = (blockIdx.x * 4 + (wid & 3)) * 64;      // < 32000
  v8f acc0 = zero8(), acc1 = zero8(), acc2 = zero8(), acc3 = zero8();
#pragma unroll
  for (int kk = 0; kk < KP; kk += 32) {
    v16bf a  = load_a_frag(Hd + (long long)m0 * KP + kk, KP, lane);
    v16bf b0 = load_b_frag(Wo + (long long)(n0     ) * KP + kk, KP, lane);
    v16bf b1 = load_b_frag(Wo + (long long)(n0 + 16) * KP + kk, KP, lane);
    v16bf b2 = load_b_frag(Wo + (long long)(n0 + 32) * KP + kk, KP, lane);
    v16bf b3 = load_b_frag(Wo + (long long)(n0 + 48) * KP + kk, KP, lane);
    acc0 = wmma_bf16(a, b0, acc0);
    acc1 = wmma_bf16(a, b1, acc1);
    acc2 = wmma_bf16(a, b2, acc2);
    acc3 = wmma_bf16(a, b3, acc3);
  }
  int half8 = (lane >> 4) << 3;
  int ln = lane & 15;
#pragma unroll
  for (int j = 0; j < 4; ++j) {
    v8f acc = (j == 0) ? acc0 : (j == 1) ? acc1 : (j == 2) ? acc2 : acc3;
    int col = n0 + j * 16 + ln;
    float bias = bout[col];
#pragma unroll
    for (int r = 0; r < 8; ++r) {
      __builtin_nontemporal_store(acc[r] + bias,
                                  out + (long long)(m0 + r + half8) * VV + col);
    }
  }
}

// ---------------- launcher ----------------
extern "C" void kernel_launch(void* const* d_in, const int* in_sizes, int n_in,
                              void* d_out, int out_size, void* d_ws, size_t ws_size,
                              hipStream_t stream) {
  const int*   x    = (const int*)d_in[0];
  const int*   y    = (const int*)d_in[1];
  const float* h0   = (const float*)d_in[2];
  const float* EN   = (const float*)d_in[3];
  const float* ZH   = (const float*)d_in[4];
  const float* eWih = (const float*)d_in[5];
  const float* eWhh = (const float*)d_in[6];
  const float* ebih = (const float*)d_in[7];
  const float* ebhh = (const float*)d_in[8];
  const float* dWih = (const float*)d_in[9];
  const float* dWhh = (const float*)d_in[10];
  const float* dbih = (const float*)d_in[11];
  const float* dbhh = (const float*)d_in[12];
  const float* Wout = (const float*)d_in[13];
  const float* bout = (const float*)d_in[14];
  float* out = (float*)d_out;

  char* base = (char*)d_ws;
  size_t off = 0;
  auto alloc = [&](size_t bytes) -> char* {
    char* p = base + off;
    off = (off + bytes + 255) & ~(size_t)255;
    return p;
  };
  bf16*  Aenc = (bf16*)alloc((size_t)MENC * KP * 2);
  bf16*  Adec = (bf16*)alloc((size_t)MDEC * KP * 2);
  bf16*  Wihe = (bf16*)alloc((size_t)HP * KP * 2);
  bf16*  Whhe = (bf16*)alloc((size_t)HP * KP * 2);
  bf16*  Wihd = (bf16*)alloc((size_t)HP * KP * 2);
  bf16*  Wob  = (bf16*)alloc((size_t)VV * KP * 2);
  float* Xenc = (float*)alloc((size_t)MENC * HP * 4);
  float* cbuf = (float*)alloc((size_t)32 * HP * 4);
  float* term = (float*)alloc((size_t)32 * HP * 4);
  bf16*  Hdec = (bf16*)alloc((size_t)MDEC * KP * 2);
  (void)ws_size; (void)in_sizes; (void)n_in; (void)out_size;

  // weight / embedding conversion to bf16 (zero-padded tiles)
  k_convert_weight<<<(HP * KP + 255) / 256, 256, 0, stream>>>(eWih, Wihe, HH, HP);
  k_convert_weight<<<(HP * KP + 255) / 256, 256, 0, stream>>>(eWhh, Whhe, HH, HP);
  k_convert_weight<<<(HP * KP + 255) / 256, 256, 0, stream>>>(dWih, Wihd, HH, HP);
  k_convert_weight<<<((long long)VV * KP + 255) / 256, 256, 0, stream>>>(Wout, Wob, VV, VV);
  k_gather_embed<<<((long long)MENC * KP + 255) / 256, 256, 0, stream>>>(x, EN, Aenc, MENC);
  k_gather_embed<<<((long long)MDEC * KP + 255) / 256, 256, 0, stream>>>(y, ZH, Adec, MDEC);

  // encoder input GEMM (hoisted out of the recurrence), scattered to [t][b] layout
  k_gemm_enc_x<<<(MENC / 16) * (HP / 16) / 8, 256, 0, stream>>>(Aenc, Wihe, ebih, ebhh, Xenc);

  // sequential encoder recurrence: one persistent workgroup, LDS ping-pong state
  k_encoder_scan<<<1, 256, 0, stream>>>(Whhe, h0, Xenc, cbuf);

  // decoder shared state term
  k_dec_term<<<(32 * HP + 255) / 256, 256, 0, stream>>>(cbuf, dWhh, dbhh, dbih, term);

  // decoder hidden GEMM + tanh -> bf16
  k_gemm_dec_h<<<(MDEC / 16) * (HP / 16) / 8, 256, 0, stream>>>(Adec, Wihd, term, Hdec);
  k_pad_hdec<<<(MDEC * (KP - HP) + 255) / 256, 256, 0, stream>>>(Hdec);

  // dominant logits GEMM: [2048 x 320] x [320 x 32000] -> 262MB f32 (store-BW bound)
  k_gemm_logits<<<dim3(VV / 256, MDEC / 32), 256, 0, stream>>>(Hdec, Wob, bout, out);
}
